// QuantizedLinear_18210661335427
// MI455X (gfx1250) — matrix operand order repaired and checked by reading, not compiled, not run on hardware
//
#include <hip/hip_runtime.h>

typedef __attribute__((ext_vector_type(16))) _Float16 v16h;
typedef __attribute__((ext_vector_type(8)))  _Float16 v8h;
typedef __attribute__((ext_vector_type(8)))  float    v8f;
typedef __attribute__((ext_vector_type(4)))  float    v4f;
typedef __attribute__((ext_vector_type(4)))  int      v4i;

#define BM 128          // workgroup tile M
#define BN 256          // workgroup tile N
#define BK 32           // K chunk (one f16 WMMA deep)
#define LSTRIDE 40      // halfs per LDS row: 32 data + 8 pad (80 B) -> bank spread
#define ABUF (BM * LSTRIDE)         // 5120 halfs
#define BBUF (BN * LSTRIDE)         // 10240 halfs
#define BUFH (ABUF + BBUF)          // 15360 halfs per buffer (30720 B)

// D = (f32)WMMA_f16(A=inp, B=qw) * scale[n] + bias[n]
// 256 threads = 8 waves, wave grid 2(M) x 4(N), each wave 64x64 = 4x4 tiles of
// v_wmma_f32_16x16x32_f16 (16 WMMAs per K chunk, one barrier per chunk via
// double-buffered LDS).
__global__ __launch_bounds__(256) void qlinear_wmma_f16_v2(
    const float* __restrict__ inp, const int* __restrict__ qw,
    const float* __restrict__ scale, const float* __restrict__ bias,
    float* __restrict__ out, int M, int N, int K)
{
  __shared__ __align__(16) _Float16 lds[2 * BUFH];   // 61440 B (of 320 KB/WGP)

  const int tid   = threadIdx.x;
  const int lane  = tid & 31;
  const int wv    = tid >> 5;
  const int wM    = wv & 1;            // 2 waves across M
  const int wN    = wv >> 1;           // 4 waves across N
  const int mBase = blockIdx.y * BM;
  const int nBase = blockIdx.x * BN;

  // staging: per thread 1 A half-row (16 f32) + 2 B half-rows (16 i32 each)
  const int srow = tid >> 1;           // 0..127
  const int sseg = (tid & 1) * 16;     // half-offset 0 or 16 within the K chunk

  const float* aPtr  = inp + (size_t)(mBase + srow)       * K + sseg;
  const int*   b0Ptr = qw  + (size_t)(nBase + srow)       * K + sseg;
  const int*   b1Ptr = qw  + (size_t)(nBase + 128 + srow) * K + sseg;

  v8f acc[4][4] = {};

  v4f aReg[4];
  v4i bReg0[4], bReg1[4];
#pragma unroll
  for (int t = 0; t < 4; ++t) {        // prefetch chunk 0 into registers
    aReg[t]  = *(const v4f*)(aPtr  + t * 4);
    bReg0[t] = *(const v4i*)(b0Ptr + t * 4);
    bReg1[t] = *(const v4i*)(b1Ptr + t * 4);
  }

  const int NK    = K / BK;
  const int chunk = lane >> 4;         // 0: lanes 0-15, 1: lanes 16-31
  const int r15   = lane & 15;

  // ---- prologue: convert chunk 0 and store into buffer 0 ----
  {
    _Float16* Ah = lds;
    _Float16* Bh = lds + ABUF;
    v8h pa0, pa1, pb00, pb01, pb10, pb11;
#pragma unroll
    for (int t = 0; t < 2; ++t)
#pragma unroll
      for (int e = 0; e < 4; ++e) {
        pa0[t * 4 + e]  = (_Float16)aReg[t][e];
        pa1[t * 4 + e]  = (_Float16)aReg[t + 2][e];
        pb00[t * 4 + e] = (_Float16)(float)bReg0[t][e];
        pb01[t * 4 + e] = (_Float16)(float)bReg0[t + 2][e];
        pb10[t * 4 + e] = (_Float16)(float)bReg1[t][e];
        pb11[t * 4 + e] = (_Float16)(float)bReg1[t + 2][e];
      }
    *(v8h*)(Ah + srow * LSTRIDE + sseg)             = pa0;
    *(v8h*)(Ah + srow * LSTRIDE + sseg + 8)         = pa1;
    *(v8h*)(Bh + srow * LSTRIDE + sseg)             = pb00;
    *(v8h*)(Bh + srow * LSTRIDE + sseg + 8)         = pb01;
    *(v8h*)(Bh + (128 + srow) * LSTRIDE + sseg)     = pb10;
    *(v8h*)(Bh + (128 + srow) * LSTRIDE + sseg + 8) = pb11;
  }

  for (int kc = 0; kc < NK; ++kc) {
    __syncthreads();   // buffer kc&1 now readable; buffer (kc+1)&1 now writable

    // issue next chunk's global loads; latency hidden under the 16 WMMAs
    if (kc + 1 < NK) {
      const float* ap  = aPtr  + (size_t)(kc + 1) * BK;
      const int*   bp0 = b0Ptr + (size_t)(kc + 1) * BK;
      const int*   bp1 = b1Ptr + (size_t)(kc + 1) * BK;
#pragma unroll
      for (int t = 0; t < 4; ++t) {
        aReg[t]  = *(const v4f*)(ap  + t * 4);
        bReg0[t] = *(const v4i*)(bp0 + t * 4);
        bReg1[t] = *(const v4i*)(bp1 + t * 4);
      }
    }

    const _Float16* Ah = lds + (kc & 1) * BUFH;
    const _Float16* Bh = Ah + ABUF;

    // B fragments (lane = column N; lanes 0-15 K0-15, lanes 16-31 K16-31)
    v16h bF[4];
#pragma unroll
    for (int j = 0; j < 4; ++j) {
      const int row = wN * 64 + j * 16 + r15;
      v8h lo = *(const v8h*)(Bh + row * LSTRIDE + chunk * 16);
      v8h hi = *(const v8h*)(Bh + row * LSTRIDE + chunk * 16 + 8);
      bF[j] = __builtin_shufflevector(lo, hi, 0,1,2,3,4,5,6,7,8,9,10,11,12,13,14,15);
    }
    // A fragments one at a time (lanes 0-15: K0-15; 16-31: K16-31)
#pragma unroll
  for (int i = 0; i < 4; ++i) {
      const int row = wM * 64 + i * 16 + r15;
      v8h lo = *(const v8h*)(Ah + row * LSTRIDE + chunk * 16);
      v8h hi = *(const v8h*)(Ah + row * LSTRIDE + chunk * 16 + 8);
      v16h aFi = __builtin_shufflevector(lo, hi, 0,1,2,3,4,5,6,7,8,9,10,11,12,13,14,15);
#pragma unroll
      for (int j = 0; j < 4; ++j)
        acc[i][j] = __builtin_amdgcn_wmma_f32_16x16x32_f16(
            false, aFi, false, bF[j], (short)0, acc[i][j], false, false);
    }

    // convert the prefetched chunk and store to the other buffer
    if (kc + 1 < NK) {
      _Float16* Aw = lds + ((kc + 1) & 1) * BUFH;
      _Float16* Bw = Aw + ABUF;
      v8h pa0, pa1, pb00, pb01, pb10, pb11;
#pragma unroll
      for (int t = 0; t < 2; ++t)
#pragma unroll
        for (int e = 0; e < 4; ++e) {
          pa0[t * 4 + e]  = (_Float16)aReg[t][e];
          pa1[t * 4 + e]  = (_Float16)aReg[t + 2][e];
          pb00[t * 4 + e] = (_Float16)(float)bReg0[t][e];
          pb01[t * 4 + e] = (_Float16)(float)bReg0[t + 2][e];
          pb10[t * 4 + e] = (_Float16)(float)bReg1[t][e];
          pb11[t * 4 + e] = (_Float16)(float)bReg1[t + 2][e];
        }
      *(v8h*)(Aw + srow * LSTRIDE + sseg)             = pa0;
      *(v8h*)(Aw + srow * LSTRIDE + sseg + 8)         = pa1;
      *(v8h*)(Bw + srow * LSTRIDE + sseg)             = pb00;
      *(v8h*)(Bw + srow * LSTRIDE + sseg + 8)         = pb01;
      *(v8h*)(Bw + (128 + srow) * LSTRIDE + sseg)     = pb10;
      *(v8h*)(Bw + (128 + srow) * LSTRIDE + sseg + 8) = pb11;
    }
  }

  // ---- epilogue: D layout -> lane column fixed; scale/bias once per N tile ----
#pragma unroll
  for (int j = 0; j < 4; ++j) {
    const int col = nBase + wN * 64 + j * 16 + r15;
    const float s  = scale[col];
    const float bz = bias[col];
#pragma unroll
    for (int i = 0; i < 4; ++i) {
      const int m0 = mBase + wM * 64 + i * 16 + chunk * 8;  // VGPR r -> row m0+r
#pragma unroll
      for (int r = 0; r < 8; ++r)
        out[(size_t)(m0 + r) * N + col] = acc[i][j][r] * s + bz;
    }
  }
}

extern "C" void kernel_launch(void* const* d_in, const int* in_sizes, int n_in,
                              void* d_out, int out_size, void* d_ws, size_t ws_size,
                              hipStream_t stream) {
  const float* inp   = (const float*)d_in[0];
  const int*   qw    = (const int*)d_in[1];
  const float* scale = (const float*)d_in[2];
  const float* bias  = (const float*)d_in[3];
  float*       out   = (float*)d_out;

  const long Nout = in_sizes[2];                  // OUT = 16384
  const long K    = (long)in_sizes[1] / Nout;     // IN  = 4096
  const long M    = (long)in_sizes[0] / K;        // B*S = 8192

  dim3 grid((unsigned)(Nout / BN), (unsigned)(M / BM));
  qlinear_wmma_f16_v2<<<grid, 256, 0, stream>>>(inp, qw, scale, bias, out,
                                                (int)M, (int)Nout, (int)K);
}
